// TreeLstmDecoderComplete_39376260170331
// MI455X (gfx1250) — compile-verified, matrix operand order checked by reading, not run on hardware
//
#include <hip/hip_runtime.h>

// ---------------------------------------------------------------------------
// TreeLSTM decoder loss for MI455X (gfx1250, wave32, WMMA bf16 path).
// GEMMs: v_wmma_f32_16x16x32_bf16, weights async-staged to LDS with a
// 2-deep pipeline (global_load_async_to_lds_b128 + s_wait_asynccnt).
// ---------------------------------------------------------------------------

typedef __bf16 bf16_t;
typedef __attribute__((ext_vector_type(16))) __bf16 v16bf;
typedef __attribute__((ext_vector_type(8)))  __bf16 v8bf;
typedef __attribute__((ext_vector_type(8)))  float  v8f;

#define HDIM  256
#define MROWS 4096
#define HALFM 2048
#define LVLS  24
#define BROOT 32
#define VRES  97
#define VRESP 112          // 97 padded to multiple of 16
#define VNAME 2000

__device__ __forceinline__ float sigf(float x) { return 1.0f / (1.0f + __expf(-x)); }
__device__ __forceinline__ float softplusf(float x) {
    return fmaxf(x, 0.0f) + log1pf(__expf(-fabsf(x)));
}

// ---------------------------------------------------------------------------
// Generic bf16 WMMA GEMM:  C[M,N] = epi( A[M,K] * W[N,K]^T + bias[N] )
// Block: 256 threads = 8 waves; wave tile 16(M) x 64(N); block tile 32 x 256.
// W tiles [256 x 32k] double-buffered in LDS via async-to-LDS DMA; each
// thread owns one 64B row-chunk (4 x b128). A fragments loaded from global
// in the ISA 16-bit A layout (two 16B chunks per lane).
// EPI 0: store f32. EPI 1: tanh -> f32 + bf16 mirror.
// EPI 2: g=sigmoid(v); store bf16( E0 + g*E1 )   (AddGate / parent_upd)
// ---------------------------------------------------------------------------
#define BN   256
#define LDSB 40   // bf16 elems per LDS row (80B, keeps 16B alignment)

template <int EPI>
__global__ __launch_bounds__(256) void k_wmma_gemm(
    const bf16_t* __restrict__ A, int lda,
    const bf16_t* __restrict__ W, int ldw, int Ncols,
    const float* __restrict__ bias, int K,
    float* __restrict__ Cf, int ldc,
    bf16_t* __restrict__ Cb, int ldcb,
    const float* __restrict__ E0, const float* __restrict__ E1)
{
    __shared__ bf16_t sB[2][BN * LDSB];

    const int lane = threadIdx.x & 31;
    const int wave = threadIdx.x >> 5;
    const int wm = wave >> 2;              // 0..1  (M sub-tile)
    const int wn = wave & 3;               // 0..3  (N sub-tile)
    const int hi = lane >> 4;              // lane group (K-half select)
    const int lm = lane & 15;              // row (A) / col (B,D) within tile
    const int m0 = blockIdx.y * 32 + wm * 16;
    const int nBlock = blockIdx.x * BN;

    v8f acc[4] = {};

    const bf16_t* Arow = A + (size_t)(m0 + lm) * lda;

    // per-thread B staging: thread t fills LDS row t (one N-row, 32 k's)
    const int n = threadIdx.x;
    const int gn = nBlock + n;
    const bool nvalid = (gn < Ncols);
    const bf16_t* Wrow = W + (size_t)gn * ldw;
    const int nk = K >> 5;                 // K / 32 stages

    auto issue_stage = [&](int stage) {
        bf16_t* dst = &sB[stage & 1][n * LDSB];
        if (nvalid) {
            unsigned long long ga =
                (unsigned long long)(uintptr_t)(Wrow + (stage << 5));
            // generic LDS pointer carries the wave-relative offset in [31:0]
            unsigned int ld = (unsigned int)(uintptr_t)dst;
            asm volatile(
                "global_load_async_to_lds_b128 %0, %1, off\n\t"
                "global_load_async_to_lds_b128 %0, %1, off offset:16\n\t"
                "global_load_async_to_lds_b128 %0, %1, off offset:32\n\t"
                "global_load_async_to_lds_b128 %0, %1, off offset:48"
                :: "v"(ld), "v"(ga) : "memory");
        } else {
            v8bf zv = {};
            *(v8bf*)(dst + 0)  = zv;
            *(v8bf*)(dst + 8)  = zv;
            *(v8bf*)(dst + 16) = zv;
            *(v8bf*)(dst + 24) = zv;
        }
    };

    issue_stage(0);

    for (int s = 0; s < nk; ++s) {
        const int k0 = s << 5;
        // pipeline: kick off next tile, then wait for current tile's 4 loads
        if (s + 1 < nk) {
            issue_stage(s + 1);
            asm volatile("s_wait_asynccnt 4" ::: "memory");
        } else {
            asm volatile("s_wait_asynccnt 0" ::: "memory");
        }
        __syncthreads();

        // --- A fragment (16x32 bf16, ISA layout): elems 0..7 <- K=hi*8+e,
        //     elems 8..15 <- K=16+hi*8+e
        v16bf a;
        {
            v8bf a0 = *(const v8bf*)(Arow + k0 + hi * 8);
            v8bf a1 = *(const v8bf*)(Arow + k0 + 16 + hi * 8);
#pragma unroll
            for (int e = 0; e < 8; ++e) { a[e] = a0[e]; a[e + 8] = a1[e]; }
            if (k0 + 32 < K) __builtin_prefetch(Arow + k0 + 32, 0, 1);
        }

        // --- 4 WMMAs: B fragments from LDS (col lm of each 16-wide tile)
        const bf16_t* sbuf = sB[s & 1];
#pragma unroll
        for (int j = 0; j < 4; ++j) {
            const bf16_t* bp = sbuf + (wn * 64 + j * 16 + lm) * LDSB;
            v8bf b0 = *(const v8bf*)(bp + hi * 8);
            v8bf b1 = *(const v8bf*)(bp + 16 + hi * 8);
            v16bf b;
#pragma unroll
            for (int e = 0; e < 8; ++e) { b[e] = b0[e]; b[e + 8] = b1[e]; }
            acc[j] = __builtin_amdgcn_wmma_f32_16x16x32_bf16(
                false, a, false, b, (short)0, acc[j], false, false);
        }
        __syncthreads();
    }

    // --- epilogue: D layout = VGPR r, m = m0 + hi*8 + r, n = tile + lm
#pragma unroll
    for (int j = 0; j < 4; ++j) {
        const int nn = nBlock + wn * 64 + j * 16 + lm;
        if (nn >= Ncols) continue;
        const float bb = bias ? bias[nn] : 0.0f;
#pragma unroll
        for (int r = 0; r < 8; ++r) {
            const int m = m0 + hi * 8 + r;
            float v = acc[j][r] + bb;
            if constexpr (EPI == 0) {
                Cf[(size_t)m * ldc + nn] = v;
            } else if constexpr (EPI == 1) {
                float t = tanhf(v);
                Cf[(size_t)m * ldc + nn] = t;
                Cb[(size_t)m * ldcb + nn] = (bf16_t)t;
            } else {
                float g = sigf(v);
                float pu = E0[(size_t)m * HDIM + nn] + g * E1[(size_t)m * HDIM + nn];
                Cb[(size_t)m * ldcb + nn] = (bf16_t)pu;
            }
        }
    }
}

// ---------------------------------------------------------------------------
// Gather previous-level h states, build bf16 GEMM operands + embeddings.
// ---------------------------------------------------------------------------
__global__ void k_gather(const int* __restrict__ pidx, const int* __restrict__ sidx,
                         int sub,
                         const float* __restrict__ prev_hp, const float* __restrict__ prev_hs,
                         float* __restrict__ g_hp, float* __restrict__ g_hs,
                         bf16_t* __restrict__ ABp,   // [M,512] = [h_p | h_s]
                         bf16_t* __restrict__ Xp,    // [M,512] = [emb | parent_upd(later)]
                         bf16_t* __restrict__ Xs,    // [M,512] = [emb | h_s]
                         const bf16_t* __restrict__ emb16,
                         const int* __restrict__ featl)
{
    const int i = blockIdx.x;
    const int h = threadIdx.x;
    const int pi = pidx[i] - sub;
    const int si = sidx[i] - sub;
    float vp = prev_hp[(size_t)pi * HDIM + h];
    float vs = prev_hs[(size_t)si * HDIM + h];
    g_hp[(size_t)i * HDIM + h] = vp;
    g_hs[(size_t)i * HDIM + h] = vs;
    ABp[(size_t)i * 512 + h] = (bf16_t)vp;
    ABp[(size_t)i * 512 + HDIM + h] = (bf16_t)vs;
    bf16_t e = emb16[(size_t)featl[i] * HDIM + h];
    Xp[(size_t)i * 512 + h] = e;
    Xs[(size_t)i * 512 + h] = e;
    Xs[(size_t)i * 512 + HDIM + h] = (bf16_t)vs;
}

// ---------------------------------------------------------------------------
// Topology BCE heads (3 x H->1), per-block partial sums (deterministic).
// ---------------------------------------------------------------------------
__global__ void k_topo(const float* __restrict__ hpred,
                       const float* __restrict__ wd, const float* __restrict__ bd,
                       const float* __restrict__ ww, const float* __restrict__ bw,
                       const float* __restrict__ wr, const float* __restrict__ br,
                       const float* __restrict__ isp, const float* __restrict__ hsib,
                       const float* __restrict__ isr, float* __restrict__ partial)
{
    const int row = blockIdx.x * blockDim.x + threadIdx.x;
    const float* h = hpred + (size_t)row * HDIM;
    float d = 0.0f, w = 0.0f, r = 0.0f;
    for (int k = 0; k < HDIM; ++k) {
        float x = h[k];
        d += x * wd[k]; w += x * ww[k]; r += x * wr[k];
    }
    d += bd[0]; w += bw[0]; r += br[0];
    float lo = (softplusf(d) - d * isp[row]) +
               (softplusf(w) - w * hsib[row]) +
               (softplusf(r) - r * isr[row]);
    __shared__ float s[256];
    s[threadIdx.x] = lo;
    __syncthreads();
    for (int st = 128; st > 0; st >>= 1) {
        if (threadIdx.x < st) s[threadIdx.x] += s[threadIdx.x + st];
        __syncthreads();
    }
    if (threadIdx.x == 0) partial[blockIdx.x] = s[0];
}

// ---------------------------------------------------------------------------
// Cross-entropy per row (one 256-thread block per row): lse - picked.
// ---------------------------------------------------------------------------
__global__ void k_ce(const float* __restrict__ logits, int ldc, int V,
                     const int* __restrict__ labels, int labOff,
                     float* __restrict__ partial)
{
    const int row = blockIdx.x;
    const float* Lr = logits + (size_t)row * ldc;
    __shared__ float s[256];
    float mx = -INFINITY;
    for (int c = threadIdx.x; c < V; c += 256) mx = fmaxf(mx, Lr[c]);
    s[threadIdx.x] = mx;
    __syncthreads();
    for (int st = 128; st > 0; st >>= 1) {
        if (threadIdx.x < st) s[threadIdx.x] = fmaxf(s[threadIdx.x], s[threadIdx.x + st]);
        __syncthreads();
    }
    mx = s[0];
    __syncthreads();
    float sum = 0.0f;
    for (int c = threadIdx.x; c < V; c += 256) sum += __expf(Lr[c] - mx);
    s[threadIdx.x] = sum;
    __syncthreads();
    for (int st = 128; st > 0; st >>= 1) {
        if (threadIdx.x < st) s[threadIdx.x] += s[threadIdx.x + st];
        __syncthreads();
    }
    if (threadIdx.x == 0) {
        int lab = labels[row] - labOff;
        partial[row] = mx + __logf(s[0]) - Lr[lab];
    }
}

// ---------------------------------------------------------------------------
// LSTM pointwise cell update; gathers prev c directly; writes current level.
// ---------------------------------------------------------------------------
__global__ void k_lstm(const float* __restrict__ gp, const float* __restrict__ gs,
                       const int* __restrict__ pidx, const int* __restrict__ sidx,
                       int sub,
                       const float* __restrict__ prev_cp, const float* __restrict__ prev_cs,
                       float* __restrict__ hp, float* __restrict__ cp,
                       float* __restrict__ hs, float* __restrict__ cs)
{
    const int m = blockIdx.x;
    const int h = threadIdx.x;
    const int pi = pidx[m] - sub;
    const int si = sidx[m] - sub;
    {
        const float* g = gp + (size_t)m * 1024;
        float i_ = sigf(g[h]);
        float f_ = sigf(g[256 + h]);
        float gg = tanhf(g[512 + h]);
        float o_ = sigf(g[768 + h]);
        float c2 = f_ * prev_cp[(size_t)pi * HDIM + h] + i_ * gg;
        hp[(size_t)m * HDIM + h] = o_ * tanhf(c2);
        cp[(size_t)m * HDIM + h] = c2;
    }
    {
        const float* g = gs + (size_t)m * 1024;
        float i_ = sigf(g[h]);
        float f_ = sigf(g[256 + h]);
        float gg = tanhf(g[512 + h]);
        float o_ = sigf(g[768 + h]);
        float c2 = f_ * prev_cs[(size_t)si * HDIM + h] + i_ * gg;
        hs[(size_t)m * HDIM + h] = o_ * tanhf(c2);
        cs[(size_t)m * HDIM + h] = c2;
    }
}

// ---------------------------------------------------------------------------
// Prep / utility kernels
// ---------------------------------------------------------------------------
__global__ void k_zero_f32(float* p, int n) {
    int i = blockIdx.x * 256 + threadIdx.x;
    if (i < n) p[i] = 0.0f;
}
__global__ void k_f2b(const float* __restrict__ s, bf16_t* __restrict__ d, int n) {
    int i = blockIdx.x * 256 + threadIdx.x;
    if (i < n) d[i] = (bf16_t)s[i];
}
__global__ void k_pack2(const float* __restrict__ A, const float* __restrict__ B,
                        bf16_t* __restrict__ d, int O, int Kh) {
    int i = blockIdx.x * 256 + threadIdx.x;
    int tot = O * 2 * Kh;
    if (i < tot) {
        int o = i / (2 * Kh);
        int k = i % (2 * Kh);
        float v = (k < Kh) ? A[(size_t)o * Kh + k] : B[(size_t)o * Kh + (k - Kh)];
        d[i] = (bf16_t)v;
    }
}
__global__ void k_padw(const float* __restrict__ s, bf16_t* __restrict__ d,
                       int rin, int cols, int rout) {
    int i = blockIdx.x * 256 + threadIdx.x;
    int tot = rout * cols;
    if (i < tot) {
        int r = i / cols;
        d[i] = (r < rin) ? (bf16_t)s[i] : (bf16_t)0.0f;
    }
}
__global__ void k_vadd(const float* __restrict__ a, const float* __restrict__ b,
                       float* __restrict__ d, int n) {
    int i = blockIdx.x * 256 + threadIdx.x;
    if (i < n) d[i] = a[i] + b[i];
}
__global__ void k_padb(const float* __restrict__ s, float* __restrict__ d,
                       int nin, int nout) {
    int i = blockIdx.x * 256 + threadIdx.x;
    if (i < nout) d[i] = (i < nin) ? s[i] : 0.0f;
}
__global__ void k_reduce(const float* __restrict__ partial, int n, float* __restrict__ out) {
    __shared__ float s[256];
    float acc = 0.0f;
    for (int i = threadIdx.x; i < n; i += 256) acc += partial[i];
    s[threadIdx.x] = acc;
    __syncthreads();
    for (int st = 128; st > 0; st >>= 1) {
        if (threadIdx.x < st) s[threadIdx.x] += s[threadIdx.x + st];
        __syncthreads();
    }
    if (threadIdx.x == 0) out[0] = s[0];
}

// ---------------------------------------------------------------------------
extern "C" void kernel_launch(void* const* d_in, const int* in_sizes, int n_in,
                              void* d_out, int out_size, void* d_ws, size_t ws_size,
                              hipStream_t stream)
{
    (void)in_sizes; (void)n_in; (void)out_size; (void)ws_size;

    const float* z           = (const float*)d_in[0];
    const int*   parent_idx  = (const int*)  d_in[1];
    const int*   sib_idx     = (const int*)  d_in[2];
    /* d_in[3] level_start: value is B + l*M by construction, computed on host */
    const int*   feat        = (const int*)  d_in[4];
    const float* is_parent   = (const float*)d_in[5];
    const float* has_sibling = (const float*)d_in[6];
    const float* is_res      = (const float*)d_in[7];
    const float* W_l2h = (const float*)d_in[8];
    const float* b_l2h = (const float*)d_in[9];
    const float* Wih_p = (const float*)d_in[10];
    const float* Whh_p = (const float*)d_in[11];
    const float* bih_p = (const float*)d_in[12];
    const float* bhh_p = (const float*)d_in[13];
    const float* Wih_s = (const float*)d_in[14];
    const float* Whh_s = (const float*)d_in[15];
    const float* bih_s = (const float*)d_in[16];
    const float* bhh_s = (const float*)d_in[17];
    const float* W1 = (const float*)d_in[18];
    const float* b1 = (const float*)d_in[19];
    const float* W2 = (const float*)d_in[20];
    const float* b2 = (const float*)d_in[21];
    const float* Wg = (const float*)d_in[22];
    const float* bg = (const float*)d_in[23];
    const float* wd = (const float*)d_in[24];
    const float* bd = (const float*)d_in[25];
    const float* ww = (const float*)d_in[26];
    const float* bw = (const float*)d_in[27];
    const float* wr = (const float*)d_in[28];
    const float* br = (const float*)d_in[29];
    const float* Wres  = (const float*)d_in[30];
    const float* bres  = (const float*)d_in[31];
    const float* Wname = (const float*)d_in[32];
    const float* bname = (const float*)d_in[33];
    const float* embt  = (const float*)d_in[34];

    // ---- workspace carve (all 256B aligned) -------------------------------
    char* base = (char*)d_ws;
    size_t off = 0;
    auto carve = [&](size_t bytes) -> char* {
        char* p = base + off;
        off += (bytes + 255) & ~(size_t)255;
        return p;
    };
    float* stA[4]; float* stB[4];
    for (int i = 0; i < 4; ++i) stA[i] = (float*)carve((size_t)MROWS * HDIM * 4);
    for (int i = 0; i < 4; ++i) stB[i] = (float*)carve((size_t)MROWS * HDIM * 4);
    float*  g_hp   = (float*) carve((size_t)MROWS * HDIM * 4);
    float*  g_hs   = (float*) carve((size_t)MROWS * HDIM * 4);
    bf16_t* ABp    = (bf16_t*)carve((size_t)MROWS * 512 * 2);
    bf16_t* Xp     = (bf16_t*)carve((size_t)MROWS * 512 * 2);
    bf16_t* Xs     = (bf16_t*)carve((size_t)MROWS * 512 * 2);
    float*  hpredf = (float*) carve((size_t)MROWS * HDIM * 4);
    bf16_t* hpredb = (bf16_t*)carve((size_t)MROWS * HDIM * 2);
    float*  gatesp = (float*) carve((size_t)MROWS * 1024 * 4);
    float*  gatess = (float*) carve((size_t)MROWS * 1024 * 4);
    float*  lg_name = gatesp;   // alias: logits dead before gates written
    float*  lg_res  = gatess;
    bf16_t* z16     = (bf16_t*)carve((size_t)BROOT * HDIM * 2);
    bf16_t* Wl2h16  = (bf16_t*)carve((size_t)HDIM * HDIM * 2);
    bf16_t* W12     = (bf16_t*)carve((size_t)HDIM * 512 * 2);
    bf16_t* Wg16    = (bf16_t*)carve((size_t)HDIM * HDIM * 2);
    bf16_t* Wp16    = (bf16_t*)carve((size_t)1024 * 512 * 2);
    bf16_t* Ws16    = (bf16_t*)carve((size_t)1024 * 512 * 2);
    bf16_t* WresP   = (bf16_t*)carve((size_t)VRESP * HDIM * 2);
    bf16_t* Wname16 = (bf16_t*)carve((size_t)VNAME * HDIM * 2);
    bf16_t* emb16   = (bf16_t*)carve((size_t)(VRES + VNAME) * HDIM * 2);
    float*  b12     = (float*) carve(HDIM * 4);
    float*  bp      = (float*) carve(1024 * 4);
    float*  bs      = (float*) carve(1024 * 4);
    float*  bresP   = (float*) carve(VRESP * 4);
    const int PART_PER_LVL = 16 + HALFM + HALFM;           // topo + ce_res + ce_name
    float*  partial = (float*) carve((size_t)LVLS * PART_PER_LVL * 4);

    auto g1 = [](int n) { return dim3((unsigned)((n + 255) / 256)); };

    // ---- one-time prep (runs every call; deterministic) -------------------
    k_zero_f32<<<g1(BROOT * HDIM), 256, 0, stream>>>(stA[1], BROOT * HDIM);
    k_zero_f32<<<g1(BROOT * HDIM), 256, 0, stream>>>(stA[2], BROOT * HDIM);
    k_zero_f32<<<g1(BROOT * HDIM), 256, 0, stream>>>(stA[3], BROOT * HDIM);
    k_f2b<<<g1(BROOT * HDIM), 256, 0, stream>>>(z, z16, BROOT * HDIM);
    k_f2b<<<g1(HDIM * HDIM), 256, 0, stream>>>(W_l2h, Wl2h16, HDIM * HDIM);
    k_f2b<<<g1(HDIM * HDIM), 256, 0, stream>>>(Wg, Wg16, HDIM * HDIM);
    k_f2b<<<g1(VNAME * HDIM), 256, 0, stream>>>(Wname, Wname16, VNAME * HDIM);
    k_f2b<<<g1((VRES + VNAME) * HDIM), 256, 0, stream>>>(embt, emb16, (VRES + VNAME) * HDIM);
    k_pack2<<<g1(HDIM * 512), 256, 0, stream>>>(W1, W2, W12, HDIM, HDIM);
    k_pack2<<<g1(1024 * 512), 256, 0, stream>>>(Wih_p, Whh_p, Wp16, 1024, HDIM);
    k_pack2<<<g1(1024 * 512), 256, 0, stream>>>(Wih_s, Whh_s, Ws16, 1024, HDIM);
    k_vadd<<<g1(HDIM), 256, 0, stream>>>(b1, b2, b12, HDIM);
    k_vadd<<<g1(1024), 256, 0, stream>>>(bih_p, bhh_p, bp, 1024);
    k_vadd<<<g1(1024), 256, 0, stream>>>(bih_s, bhh_s, bs, 1024);
    k_padw<<<g1(VRESP * HDIM), 256, 0, stream>>>(Wres, WresP, VRES, HDIM, VRESP);
    k_padb<<<g1(VRESP), 256, 0, stream>>>(bres, bresP, VRES, VRESP);

    // roots: h_p[:32] = z @ W_l2h^T + b_l2h   (M=32, N=256, K=256)
    k_wmma_gemm<0><<<dim3(1, 1), 256, 0, stream>>>(
        z16, HDIM, Wl2h16, HDIM, HDIM, b_l2h, HDIM,
        stA[0], HDIM, nullptr, 0, nullptr, nullptr);

    // ---- level loop (ping-pong state) -------------------------------------
    float* ph = stA[0]; float* pc = stA[1]; float* psh = stA[2]; float* psc = stA[3];
    float* ch = stB[0]; float* cc = stB[1]; float* csh = stB[2]; float* csc = stB[3];

    for (int l = 0; l < LVLS; ++l) {
        const int sub = (l == 0) ? 0 : (BROOT + (l - 1) * MROWS);
        const int* pidx = parent_idx + (size_t)l * MROWS;
        const int* sidx = sib_idx + (size_t)l * MROWS;
        const int* featl = feat + (size_t)l * MROWS;
        float* Plvl = partial + (size_t)l * PART_PER_LVL;

        k_gather<<<dim3(MROWS), 256, 0, stream>>>(
            pidx, sidx, sub, ph, psh, g_hp, g_hs, ABp, Xp, Xs, emb16, featl);

        // h_pred = tanh([h_p|h_s] @ [W1|W2]^T + b1+b2)     M=4096 N=256 K=512
        k_wmma_gemm<1><<<dim3(1, MROWS / 32), 256, 0, stream>>>(
            ABp, 512, W12, 512, HDIM, b12, 512,
            hpredf, HDIM, hpredb, HDIM, nullptr, nullptr);

        // gate = sigmoid(h_s @ Wg^T + bg); parent_upd -> Xp[:,256:512] (bf16)
        k_wmma_gemm<2><<<dim3(1, MROWS / 32), 256, 0, stream>>>(
            ABp + HDIM, 512, Wg16, HDIM, HDIM, bg, HDIM,
            nullptr, 0, Xp + HDIM, 512, g_hp, g_hs);

        // topology BCE heads
        k_topo<<<dim3(16), 256, 0, stream>>>(
            hpredf, wd, bd, ww, bw, wr, br,
            is_parent + (size_t)l * MROWS, has_sibling + (size_t)l * MROWS,
            is_res + (size_t)l * MROWS, Plvl);

        // RES logits (rows 0..2047)  N padded to 112
        k_wmma_gemm<0><<<dim3(1, HALFM / 32), 256, 0, stream>>>(
            hpredb, HDIM, WresP, HDIM, VRESP, bresP, HDIM,
            lg_res, VRESP, nullptr, 0, nullptr, nullptr);
        k_ce<<<dim3(HALFM), 256, 0, stream>>>(lg_res, VRESP, VRES, featl, 0, Plvl + 16);

        // NAME logits (rows 2048..4095)  N=2000
        k_wmma_gemm<0><<<dim3((VNAME + BN - 1) / BN, HALFM / 32), 256, 0, stream>>>(
            hpredb + (size_t)HALFM * HDIM, HDIM, Wname16, HDIM, VNAME, bname, HDIM,
            lg_name, VNAME, nullptr, 0, nullptr, nullptr);
        k_ce<<<dim3(HALFM), 256, 0, stream>>>(lg_name, VNAME, VNAME, featl + HALFM,
                                              VRES, Plvl + 16 + HALFM);

        // LSTM gate GEMMs: [emb|h] @ [Wih|Whh]^T + (bih+bhh)  M=4096 N=1024 K=512
        k_wmma_gemm<0><<<dim3(1024 / BN, MROWS / 32), 256, 0, stream>>>(
            Xp, 512, Wp16, 512, 1024, bp, 512,
            gatesp, 1024, nullptr, 0, nullptr, nullptr);
        k_wmma_gemm<0><<<dim3(1024 / BN, MROWS / 32), 256, 0, stream>>>(
            Xs, 512, Ws16, 512, 1024, bs, 512,
            gatess, 1024, nullptr, 0, nullptr, nullptr);

        // pointwise cell update -> current level buffers
        k_lstm<<<dim3(MROWS), 256, 0, stream>>>(
            gatesp, gatess, pidx, sidx, sub, pc, psc, ch, cc, csh, csc);

        // swap ping-pong
        float* t;
        t = ph;  ph = ch;  ch = t;
        t = pc;  pc = cc;  cc = t;
        t = psh; psh = csh; csh = t;
        t = psc; psc = csc; csc = t;
    }

    // ---- deterministic final loss reduction -------------------------------
    k_reduce<<<dim3(1), 256, 0, stream>>>(partial, LVLS * PART_PER_LVL, (float*)d_out);
}
